// SDTMMemory_9234179686571
// MI455X (gfx1250) — matrix operand order repaired and checked by reading, not compiled, not run on hardware
//
#include <hip/hip_runtime.h>
#include <hip/hip_bf16.h>
#include <math.h>

// Problem constants (match reference)
#define NTOK 16384   // B*S = 4*4096
#define DM   1024    // D_MODEL == D_MEM_TOTAL
#define NH   8
#define DMEM 128

// LDS B-tile staging: 64 rows x 32 bf16, row stride padded to 40 ushorts (80B)
// so the 16 rows hit by one ds_load_b128 land in distinct bank groups.
#define BROW 40
#define BTILE (64 * BROW)

#if defined(__HIP_DEVICE_COMPILE__) && \
    __has_builtin(__builtin_amdgcn_global_load_async_to_lds_b128) && \
    __has_builtin(__builtin_amdgcn_s_wait_asynccnt)
#define USE_ASYNC_LDS 1
#else
#define USE_ASYNC_LDS 0
#endif

typedef __attribute__((ext_vector_type(16))) __bf16        v16bf;
typedef __attribute__((ext_vector_type(8)))  float         v8f;
typedef __attribute__((ext_vector_type(4)))  unsigned int  v4u;

union FragBF {
  v4u   q[2];  // 2 x 16B = 32B = 16 bf16
  v16bf v;
};

__device__ __forceinline__ unsigned short f2bf(float f) {
  unsigned int u = __float_as_uint(f);
  if ((u & 0x7fffffffu) > 0x7f800000u) return (unsigned short)((u >> 16) | 0x40u); // NaN
  u += 0x7fffu + ((u >> 16) & 1u);  // round-to-nearest-even
  return (unsigned short)(u >> 16);
}

__device__ __forceinline__ float sigmoidf_(float v) { return 1.0f / (1.0f + expf(-v)); }
__device__ __forceinline__ float gelu_exact(float v) {
  return 0.5f * v * (1.0f + erff(v * 0.70710678118654752f));
}

// ---------------------------------------------------------------------------
// Conversions
// ---------------------------------------------------------------------------
__global__ void k_f32_to_bf16(const float* __restrict__ src,
                              unsigned short* __restrict__ dst, int n) {
  for (int i = blockIdx.x * blockDim.x + threadIdx.x; i < n;
       i += gridDim.x * blockDim.x)
    dst[i] = f2bf(src[i]);
}

// Mt[h][e][d] = M[h][d][e]  (so WMMA B-fragment columns are contiguous)
__global__ void k_transM(const float* __restrict__ M,
                         unsigned short* __restrict__ Mt) {
  const int n = NH * DMEM * DMEM;
  for (int i = blockIdx.x * blockDim.x + threadIdx.x; i < n;
       i += gridDim.x * blockDim.x) {
    int hh  = i >> 14;          // /16384
    int rem = i & 16383;
    int e   = rem >> 7;
    int d   = rem & 127;
    Mt[i] = f2bf(M[(hh << 14) + (d << 7) + e]);
  }
}

// ---------------------------------------------------------------------------
// Gate: alpha[token][h] = sigmoid(x . gate_w[h] + gate_b[h]); one wave/token
// ---------------------------------------------------------------------------
__global__ __launch_bounds__(256) void k_gate(const float* __restrict__ x,
                                              const float* __restrict__ gw,
                                              const float* __restrict__ gb,
                                              float* __restrict__ alpha) {
  const int lane  = threadIdx.x & 31;
  const int wave  = threadIdx.x >> 5;
  const int token = blockIdx.x * 8 + wave;
  const float4* xr = (const float4*)(x + (size_t)token * DM);
  float acc[NH] = {};
#pragma unroll
  for (int j = 0; j < 8; ++j) {
    float4 xv = xr[j * 32 + lane];
#pragma unroll
    for (int hh = 0; hh < NH; ++hh) {
      float4 gv = ((const float4*)(gw + (size_t)hh * DM))[j * 32 + lane];
      acc[hh] += xv.x * gv.x + xv.y * gv.y + xv.z * gv.z + xv.w * gv.w;
    }
  }
#pragma unroll
  for (int hh = 0; hh < NH; ++hh) {
#pragma unroll
    for (int off = 16; off > 0; off >>= 1)
      acc[hh] += __shfl_xor(acc[hh], off, 32);
  }
  if (lane == 0) {
#pragma unroll
    for (int hh = 0; hh < NH; ++hh)
      alpha[(size_t)token * NH + hh] = sigmoidf_(acc[hh] + gb[hh]);
  }
}

// ---------------------------------------------------------------------------
// WMMA fragment loaders (contiguous k = 16*(lane/16) + s, same order for A and B, wave32)
// ---------------------------------------------------------------------------
__device__ __forceinline__ FragBF load_a(const unsigned short* __restrict__ arow,
                                         int k, int ah) {
  FragBF a;
  a.q[0] = *(const v4u*)(arow + k + ah * 16);
  a.q[1] = *(const v4u*)(arow + k + ah * 16 + 8);
  return a;
}
__device__ __forceinline__ FragBF load_b(const unsigned short* __restrict__ wrow) {
  FragBF b;
  b.q[0] = *(const v4u*)(wrow);
  b.q[1] = *(const v4u*)(wrow + 8);
  return b;
}
__device__ __forceinline__ v8f wmma_bf16(const FragBF& a, const FragBF& b, v8f c) {
  return __builtin_amdgcn_wmma_f32_16x16x32_bf16(false, a.v, false, b.v,
                                                 (short)0, c, false, false);
}

#if USE_ASYNC_LDS
// Exact parameter types per hipcc diagnostic: first arg is
// 'int __vector(4) __device__ *' (AS1); LDS side is the AS3 analog.
typedef int v4i_ __attribute__((vector_size(16)));
typedef v4i_ __attribute__((address_space(1)))* gv4i_p;
typedef v4i_ __attribute__((address_space(3)))* lv4i_p;

__device__ __forceinline__ void stage_async(const unsigned short* __restrict__ g,
                                            unsigned short* __restrict__ l) {
  __builtin_amdgcn_global_load_async_to_lds_b128(
      (gv4i_p)(const void*)g, (lv4i_p)(void*)l, 0, 0);
}

__device__ __forceinline__ void lds_b4(const unsigned short* __restrict__ lb,
                                       FragBF b[4]) {
#pragma unroll
  for (int t = 0; t < 4; ++t) {
    b[t].q[0] = *(const v4u*)(lb + t * 16 * BROW);
    b[t].q[1] = *(const v4u*)(lb + t * 16 * BROW + 8);
  }
}
#endif

// ---------------------------------------------------------------------------
// Core 16Mx64N strip GEMM over K=DM: acc[4] += A(row0:16, :) * W(col0:64, :)^T
// Async path: LDS double buffer, K unrolled by 2 with A-register ping-pong;
// all staging addresses (2 LDS dsts + global src base) hoisted out of the
// loop so the body has no dependent VALU address math (no hazard nops).
// ---------------------------------------------------------------------------
__device__ __forceinline__ void gemm_strip(const unsigned short* __restrict__ arow,
                                           const unsigned short* __restrict__ Wb,
                                           int col0, int am, int ah,
                                           unsigned short* __restrict__ sB, // [2][BTILE]
                                           v8f acc[4]) {
#if USE_ASYNC_LDS
  // Per-thread staging addresses: thread = (row 0..63, chunk 0..3); each
  // stages 16B; one async instruction per wave per tile (256 thr x 16B = 4KB).
  const int tid    = threadIdx.x;
  const int brow_  = tid >> 2;
  const int bchunk = tid & 3;
  const unsigned short* gsrc = Wb + (size_t)(col0 + brow_) * DM + bchunk * 8;
  unsigned short* ldst0 = sB + brow_ * BROW + bchunk * 8;
  unsigned short* ldst1 = ldst0 + BTILE;

  stage_async(gsrc, ldst0);
  FragBF a0 = load_a(arow, 0, ah);
  __builtin_amdgcn_s_wait_asynccnt(0);
  __syncthreads();
  const unsigned short* lb0 = sB + am * BROW + ah * 16;
  const unsigned short* lb1 = lb0 + BTILE;
  FragBF a1;
  for (int k = 0; k < DM; k += 64) {
    // ---- half 1: compute buf0 (K=k), stage buf1 (K=k+32) ----
    stage_async(gsrc + k + 32, ldst1);
    a1 = load_a(arow, k + 32, ah);
    __builtin_prefetch(arow + k + 96, 0, 3);
    {
      FragBF b[4];
      lds_b4(lb0, b);
#pragma unroll
      for (int t = 0; t < 4; ++t) acc[t] = wmma_bf16(a0, b[t], acc[t]);
    }
    __builtin_amdgcn_s_wait_asynccnt(0);
    __syncthreads();
    // ---- half 2: compute buf1 (K=k+32), stage buf0 (K=k+64) ----
    const bool more = (k + 64) < DM;
    if (more) {
      stage_async(gsrc + k + 64, ldst0);
      a0 = load_a(arow, k + 64, ah);
    }
    {
      FragBF b[4];
      lds_b4(lb1, b);
#pragma unroll
      for (int t = 0; t < 4; ++t) acc[t] = wmma_bf16(a1, b[t], acc[t]);
    }
    __builtin_amdgcn_s_wait_asynccnt(0);
    __syncthreads();
  }
#else
  (void)sB;
  FragBF a_cur = load_a(arow, 0, ah);
  FragBF b_cur[4];
#pragma unroll
  for (int t = 0; t < 4; ++t)
    b_cur[t] = load_b(Wb + (size_t)(col0 + t * 16 + am) * DM + ah * 16);
  for (int k = 0; k < DM; k += 32) {
    const bool more = (k + 32) < DM;
    FragBF a_nxt, b_nxt[4];
    if (more) {
      a_nxt = load_a(arow, k + 32, ah);
#pragma unroll
      for (int t = 0; t < 4; ++t)
        b_nxt[t] = load_b(Wb + (size_t)(col0 + t * 16 + am) * DM + k + 32 + ah * 16);
    }
#pragma unroll
    for (int t = 0; t < 4; ++t)
      acc[t] = wmma_bf16(a_cur, b_cur[t], acc[t]);
    if (more) {
      a_cur = a_nxt;
#pragma unroll
      for (int t = 0; t < 4; ++t) b_cur[t] = b_nxt[t];
    }
  }
#endif
}

// ---------------------------------------------------------------------------
// Encode: z = gelu(x @ W_enc^T) as bf16.  grid(NTOK/128, DM/64), block 256.
// ---------------------------------------------------------------------------
__global__ __launch_bounds__(256) void k_encode(const unsigned short* __restrict__ Xb,
                                                const unsigned short* __restrict__ Wb,
                                                unsigned short* __restrict__ Zb) {
  __shared__ __align__(16) unsigned short sB[2 * BTILE];
  const int lane = threadIdx.x & 31;
  const int wave = threadIdx.x >> 5;
  const int row0 = (blockIdx.x * 8 + wave) * 16;
  const int col0 = blockIdx.y * 64;
  const int am = lane & 15, ah = lane >> 4;

  const unsigned short* arow = Xb + (size_t)(row0 + am) * DM;
  v8f acc[4] = {};
  gemm_strip(arow, Wb, col0, am, ah, sB, acc);

#pragma unroll
  for (int t = 0; t < 4; ++t) {
#pragma unroll
    for (int r = 0; r < 8; ++r) {
      const int row = row0 + r + ah * 8;
      const int col = col0 + t * 16 + am;
      Zb[(size_t)row * DM + col] = f2bf(gelu_exact(acc[t][r]));
    }
  }
}

// ---------------------------------------------------------------------------
// Decode: out = sigmoid(inject) * (r @ W_dec^T), f32 output. Same tiling.
// ---------------------------------------------------------------------------
__global__ __launch_bounds__(256) void k_decode(const unsigned short* __restrict__ Rb,
                                                const unsigned short* __restrict__ Wb,
                                                const float* __restrict__ inject,
                                                float* __restrict__ Out) {
  __shared__ __align__(16) unsigned short sB[2 * BTILE];
  const int lane = threadIdx.x & 31;
  const int wave = threadIdx.x >> 5;
  const int row0 = (blockIdx.x * 8 + wave) * 16;
  const int col0 = blockIdx.y * 64;
  const int am = lane & 15, ah = lane >> 4;
  const float s = sigmoidf_(inject[0]);

  const unsigned short* arow = Rb + (size_t)(row0 + am) * DM;
  v8f acc[4] = {};
  gemm_strip(arow, Wb, col0, am, ah, sB, acc);

#pragma unroll
  for (int t = 0; t < 4; ++t) {
#pragma unroll
    for (int r = 0; r < 8; ++r) {
      const int row = row0 + r + ah * 8;
      const int col = col0 + t * 16 + am;
      Out[(size_t)row * DM + col] = s * acc[t][r];
    }
  }
}

// ---------------------------------------------------------------------------
// Heads: rf = z_h @ M_fast[h], rs = z_h @ M_slow[h], r = a*rf + (1-a)*rs
// Wave tile 16 tokens x 64 e (fast+slow), K = 128. grid(NTOK/128, NH, 2).
// Fully unrolled K loop; all 8 B fragments loaded before the 8 WMMAs.
// ---------------------------------------------------------------------------
__global__ __launch_bounds__(256) void k_heads(const unsigned short* __restrict__ Zb,
                                               const unsigned short* __restrict__ Mtf,
                                               const unsigned short* __restrict__ Mts,
                                               const float* __restrict__ alpha,
                                               unsigned short* __restrict__ Rb) {
  const int lane = threadIdx.x & 31;
  const int wave = threadIdx.x >> 5;
  const int row0 = blockIdx.x * 128 + wave * 16;
  const int h    = blockIdx.y;
  const int e0   = blockIdx.z * 64;
  const int am = lane & 15, ah = lane >> 4;

  const unsigned short* arow = Zb + (size_t)(row0 + am) * DM + h * DMEM;
  const unsigned short* mf = Mtf + (size_t)h * DMEM * DMEM;
  const unsigned short* ms = Mts + (size_t)h * DMEM * DMEM;

  v8f accF[4] = {}, accS[4] = {};
  FragBF a_cur = load_a(arow, 0, ah);
#pragma unroll
  for (int k = 0; k < DMEM; k += 32) {
    const bool more = (k + 32) < DMEM;
    FragBF bF[4], bS[4];
#pragma unroll
    for (int t = 0; t < 4; ++t) {
      const size_t roff = (size_t)(e0 + t * 16 + am) * DMEM + k + ah * 16;
      bF[t] = load_b(mf + roff);
      bS[t] = load_b(ms + roff);
    }
    FragBF a_nxt;
    if (more) a_nxt = load_a(arow, k + 32, ah);
#pragma unroll
    for (int t = 0; t < 4; ++t)
      accF[t] = wmma_bf16(a_cur, bF[t], accF[t]);
#pragma unroll
    for (int t = 0; t < 4; ++t)
      accS[t] = wmma_bf16(a_cur, bS[t], accS[t]);
    if (more) a_cur = a_nxt;
  }

#pragma unroll
  for (int r = 0; r < 8; ++r) {
    const int row = row0 + r + ah * 8;
    const float al = alpha[(size_t)row * NH + h];
#pragma unroll
    for (int t = 0; t < 4; ++t) {
      const float v = al * accF[t][r] + (1.0f - al) * accS[t][r];
      Rb[(size_t)row * DM + h * DMEM + e0 + t * 16 + am] = f2bf(v);
    }
  }
}

// ---------------------------------------------------------------------------
extern "C" void kernel_launch(void* const* d_in, const int* in_sizes, int n_in,
                              void* d_out, int out_size, void* d_ws, size_t ws_size,
                              hipStream_t stream) {
  (void)in_sizes; (void)n_in; (void)out_size; (void)ws_size;

  const float* x      = (const float*)d_in[0];
  const float* W_enc  = (const float*)d_in[1];
  const float* W_dec  = (const float*)d_in[2];
  const float* gate_w = (const float*)d_in[3];
  const float* gate_b = (const float*)d_in[4];
  const float* M_fast = (const float*)d_in[5];
  const float* M_slow = (const float*)d_in[6];
  const float* inj    = (const float*)d_in[7];

  char* ws = (char*)d_ws;
  const size_t WENC_OFF  = 0;
  const size_t WDEC_OFF  = WENC_OFF  + (size_t)DM * DM * 2;
  const size_t MTF_OFF   = WDEC_OFF  + (size_t)DM * DM * 2;
  const size_t MTS_OFF   = MTF_OFF   + (size_t)NH * DMEM * DMEM * 2;
  const size_t ALPHA_OFF = MTS_OFF   + (size_t)NH * DMEM * DMEM * 2;
  const size_t XBF_OFF   = ALPHA_OFF + (size_t)NTOK * NH * 4;
  const size_t ZBF_OFF   = XBF_OFF   + (size_t)NTOK * DM * 2;
  const size_t RBF_OFF   = ZBF_OFF   + (size_t)NTOK * DM * 2;

  unsigned short* Wenc_b = (unsigned short*)(ws + WENC_OFF);
  unsigned short* Wdec_b = (unsigned short*)(ws + WDEC_OFF);
  unsigned short* Mtf    = (unsigned short*)(ws + MTF_OFF);
  unsigned short* Mts    = (unsigned short*)(ws + MTS_OFF);
  float*          alpha  = (float*)(ws + ALPHA_OFF);
  unsigned short* Xb     = (unsigned short*)(ws + XBF_OFF);
  unsigned short* Zb     = (unsigned short*)(ws + ZBF_OFF);
  unsigned short* Rb     = (unsigned short*)(ws + RBF_OFF);

  // Stage 0: precision conversion / transposes
  k_f32_to_bf16<<<512,  256, 0, stream>>>(W_enc, Wenc_b, DM * DM);
  k_f32_to_bf16<<<512,  256, 0, stream>>>(W_dec, Wdec_b, DM * DM);
  k_transM     <<<128,  256, 0, stream>>>(M_fast, Mtf);
  k_transM     <<<128,  256, 0, stream>>>(M_slow, Mts);
  k_f32_to_bf16<<<2048, 256, 0, stream>>>(x, Xb, NTOK * DM);

  // Stage 1: gate
  k_gate<<<NTOK / 8, 256, 0, stream>>>(x, gate_w, gate_b, alpha);

  // Stage 2: encode GEMM + GELU (WMMA bf16, async-LDS B staging)
  k_encode<<<dim3(NTOK / 128, DM / 64), 256, 0, stream>>>(Xb, Wenc_b, Zb);

  // Stage 3: per-head fast/slow reads + gated mix (WMMA bf16)
  k_heads<<<dim3(NTOK / 128, NH, 2), 256, 0, stream>>>(Zb, Mtf, Mts, alpha, Rb);

  // Stage 4: decode GEMM + inject scale (WMMA bf16 -> f32 out)
  k_decode<<<dim3(NTOK / 128, DM / 64), 256, 0, stream>>>(Rb, Wdec_b, inj, (float*)d_out);
}
